// PerPixelLinear_85263690760286
// MI455X (gfx1250) — matrix operand order repaired and checked by reading, not compiled, not run on hardware
//
#include <hip/hip_runtime.h>
#include <math.h>

typedef __attribute__((ext_vector_type(16))) _Float16 v16h;
typedef __attribute__((ext_vector_type(8)))  _Float16 v8h;
typedef __attribute__((ext_vector_type(8)))  float    v8f;

#define IN_CH   384
#define OUT_CH  384
#define HW      4096          // 64*64 pixels per image per channel
#define PIX_TILE 64
#define XN_STRIDE 392         // halfs per pixel row (384 + 8 pad)
#define WB_STRIDE 40          // halfs per out-row in staged W chunk (32 + 8 pad)
#define OUT_STRIDE 68         // floats per out-channel row (64 + 4 pad, keeps 16B align)
#define KSTEPS  12            // 384 / 32
#define NT      12            // 192 / 16 N-tiles per wave

// Dynamic LDS layout (bytes). Region A (raw x + stat scratch) is reused as the
// output-transpose buffer after the normalize phase is complete.
#define LDS_RAW   0
#define LDS_PSUM  98304                  // 256 threads * 4 floats (x2 for sumsq)
#define LDS_STAT  (98304 + 4096*2)       // mu[64], rs[64]
#define LDS_OUT   0                      // 384*68*4 = 104448 <= 107008
#define LDS_XN    107008                 // 64*392*2 = 50176
#define LDS_WBUF  (107008 + 50176)       // 2*384*40*2 = 61440
#define LDS_TOTAL (107008 + 50176 + 61440)   // 218624 bytes <= 320KB/WGP

// CDNA5 async global->LDS staging (ASYNCcnt-tracked DMA), with safe fallback.
#if defined(__gfx1250__) && __has_builtin(__builtin_amdgcn_global_load_async_to_lds_b128) && \
    __has_builtin(__builtin_amdgcn_s_wait_asynccnt)
#define USE_ASYNC_LDS 1
typedef int v4i_ __attribute__((vector_size(16)));
typedef __attribute__((address_space(1))) v4i_ g_v4i;   // global (AS1) int4
typedef __attribute__((address_space(3))) v4i_ l_v4i;   // LDS (AS3) int4
#else
#define USE_ASYNC_LDS 0
#endif

__global__ void w_to_f16_kernel(const float* __restrict__ W,
                                _Float16* __restrict__ Wh, int n) {
  int i = blockIdx.x * 256 + threadIdx.x;
  if (i < n) Wh[i] = (_Float16)W[i];
}

__global__ __launch_bounds__(256)
void ln_linear_gelu_kernel(const float* __restrict__ x,
                           const float* __restrict__ gamma,
                           const float* __restrict__ beta,
                           const _Float16* __restrict__ Wh,
                           const float* __restrict__ bias,
                           float* __restrict__ y) {
  extern __shared__ __align__(16) char smem[];
  float*    rawf  = (float*)(smem + LDS_RAW);
  float4*   raw4  = (float4*)(smem + LDS_RAW);
  float*    psumf = (float*)(smem + LDS_PSUM);           // [256][4] interleaved
  float*    psqf  = (float*)(smem + LDS_PSUM + 4096);
  float*    smu   = (float*)(smem + LDS_STAT);
  float*    srs   = (float*)(smem + LDS_STAT + 256);
  _Float16* xn    = (_Float16*)(smem + LDS_XN);
  _Float16* wbuf  = (_Float16*)(smem + LDS_WBUF);
  float*    outl  = (float*)(smem + LDS_OUT);

  const int t     = threadIdx.x;
  const int img   = blockIdx.x >> 6;
  const int qbase = (blockIdx.x & 63) * PIX_TILE;

  // ---- Phase 1: coalesced load of 64px x 384ch tile + per-pixel partial stats
  {
    const int p4 = t & 15;        // float4 column (pixels 4*p4 .. 4*p4+3)
    const int c0 = t >> 4;        // base channel
    float sx = 0.f, sy = 0.f, sz = 0.f, sw = 0.f;
    float qx = 0.f, qy = 0.f, qz = 0.f, qw = 0.f;
    #pragma unroll 4
    for (int i = 0; i < 24; ++i) {
      int c = c0 + 16 * i;
      float4 v = *(const float4*)(x + (size_t)(img * IN_CH + c) * HW + qbase + 4 * p4);
      raw4[c * 16 + p4] = v;
      sx += v.x; sy += v.y; sz += v.z; sw += v.w;
      qx += v.x * v.x; qy += v.y * v.y; qz += v.z * v.z; qw += v.w * v.w;
    }
    psumf[t * 4 + 0] = sx; psumf[t * 4 + 1] = sy; psumf[t * 4 + 2] = sz; psumf[t * 4 + 3] = sw;
    psqf [t * 4 + 0] = qx; psqf [t * 4 + 1] = qy; psqf [t * 4 + 2] = qz; psqf [t * 4 + 3] = qw;
  }
  __syncthreads();

  // ---- Phase 1b: finalize mean / rsqrt(var) for 64 pixels
  if (t < 64) {
    float mu = 0.f, m2 = 0.f;
    #pragma unroll
    for (int s = 0; s < 16; ++s) {
      mu += psumf[s * 64 + t];
      m2 += psqf [s * 64 + t];
    }
    mu *= (1.0f / IN_CH);
    float var = m2 * (1.0f / IN_CH) - mu * mu;
    smu[t] = mu;
    srs[t] = rsqrtf(var + 1e-5f);
  }
  __syncthreads();

  // ---- Phase 2: normalize + affine, convert to f16, pixel-major layout for A-frags
  {
    const int p  = t & 63;
    const int cq = t >> 6;
    float mu = smu[p], rs = srs[p];
    #pragma unroll 4
    for (int cb = 0; cb < IN_CH; cb += 4) {
      int c = cb + cq;
      float v = rawf[c * 64 + p];
      xn[p * XN_STRIDE + c] = (_Float16)((v - mu) * rs * gamma[c] + beta[c]);
    }
  }
  __syncthreads();

  // ---- Phase 3: WMMA GEMM  (xn[64x384] x W^T[384x384])
  const int lane = t & 31;
  const int wave = t >> 5;
  const int grp  = wave >> 1;                 // pixel group: pixels 16*grp..+15
  const int half = wave & 1;                  // out-channel half: 192*half..
  const int l15  = lane & 15;
  const int hi16 = (lane >> 4) & 1;

  v8f acc[NT];
  #pragma unroll
  for (int j = 0; j < NT; ++j) acc[j] = v8f{};

  // Stage one 384x32 f16 chunk of W into LDS (24 KB), double buffered.
  auto stageW = [&](int k, int buf) {
    _Float16* dst = wbuf + buf * (OUT_CH * WB_STRIDE);
    #pragma unroll
    for (int ii = 0; ii < 6; ++ii) {
      int idx = t + 256 * ii;                 // 0..1535 = 384 rows * 4 segs
      int o = idx >> 2, seg = idx & 3;
      const _Float16* src = Wh + (size_t)o * IN_CH + k * 32 + seg * 8;
      _Float16* d = dst + o * WB_STRIDE + seg * 8;
#if USE_ASYNC_LDS
      __builtin_amdgcn_global_load_async_to_lds_b128(
          (g_v4i*)src, (l_v4i*)d, 0, 0);
#else
      *(float4*)d = *(const float4*)src;
#endif
    }
  };

  union AV { v16h v; v8h h[2]; };

  stageW(0, 0);
#if USE_ASYNC_LDS
  __builtin_amdgcn_s_wait_asynccnt(0);
#endif
  __syncthreads();

  for (int k = 0; k < KSTEPS; ++k) {
    if (k + 1 < KSTEPS) stageW(k + 1, (k + 1) & 1);   // overlaps with WMMA below

    // A fragment: 16 pixels x 32 channels (ISA 16-bit A layout)
    AV a;
    {
      const _Float16* ap = xn + (16 * grp + l15) * XN_STRIDE + 32 * k + hi16 * 8;
      a.h[0] = *(const v8h*)ap;          // K = koff .. koff+7
      a.h[1] = *(const v8h*)(ap + 16);   // K = koff+16 .. koff+23
    }

    const _Float16* wb = wbuf + (k & 1) * (OUT_CH * WB_STRIDE) + hi16 * 8;
    #pragma unroll
    for (int j = 0; j < NT; ++j) {
      AV b;
      const _Float16* bp = wb + (half * 192 + 16 * j + l15) * WB_STRIDE;
      b.h[0] = *(const v8h*)bp;          // K = koff .. koff+7
      b.h[1] = *(const v8h*)(bp + 16);   // K = koff+16 .. koff+23
      acc[j] = __builtin_amdgcn_wmma_f32_16x16x32_f16(
          false, a.v, false, b.v, (short)0, acc[j], false, false);
    }
#if USE_ASYNC_LDS
    __builtin_amdgcn_s_wait_asynccnt(0);   // my async stores for chunk k+1 landed
#endif
    __syncthreads();                       // all waves' staging visible
  }

  // ---- Epilogue: bias + exact GELU, transpose through LDS (reuses raw-x region)
  #pragma unroll
  for (int j = 0; j < NT; ++j) {
    int o = half * 192 + 16 * j + l15;
    float bo = bias[o];
    float* orow = outl + o * OUT_STRIDE + 16 * grp + hi16 * 8;
    float g[8];
    #pragma unroll
    for (int r = 0; r < 8; ++r) {
      float v = acc[j][r] + bo;
      g[r] = 0.5f * v * (1.0f + erff(v * 0.70710678118654752f));
    }
    float4 lo = {g[0], g[1], g[2], g[3]};
    float4 hi = {g[4], g[5], g[6], g[7]};
    *(float4*)orow = lo;
    *(float4*)(orow + 4) = hi;
  }
  __syncthreads();

  // ---- Coalesced NCHW writeback
  #pragma unroll 4
  for (int ii = 0; ii < 24; ++ii) {
    int idx = t + 256 * ii;                  // 0..6143 = 384 rows * 16 float4
    int o = idx >> 4, p4 = idx & 15;
    float4 v = *(const float4*)(outl + o * OUT_STRIDE + 4 * p4);
    *(float4*)(y + (size_t)(img * OUT_CH + o) * HW + qbase + 4 * p4) = v;
  }
}

extern "C" void kernel_launch(void* const* d_in, const int* in_sizes, int n_in,
                              void* d_out, int out_size, void* d_ws, size_t ws_size,
                              hipStream_t stream) {
  const float* x     = (const float*)d_in[0];
  const float* gam   = (const float*)d_in[1];
  const float* bet   = (const float*)d_in[2];
  const float* W     = (const float*)d_in[3];
  const float* bias  = (const float*)d_in[4];
  float*       y     = (float*)d_out;
  _Float16*    Wh    = (_Float16*)d_ws;     // 384*384*2 = 288 KB scratch

  const int wn = OUT_CH * IN_CH;
  w_to_f16_kernel<<<(wn + 255) / 256, 256, 0, stream>>>(W, Wh, wn);

  // 32 images * 64 pixel-tiles = 2048 blocks, 256 threads (8 wave32)
  ln_linear_gelu_kernel<<<dim3(2048), dim3(256), LDS_TOTAL, stream>>>(
      x, gam, bet, Wh, bias, y);
}
